// AttentionHead_39178691674423
// MI455X (gfx1250) — compile-verified, matrix operand order checked by reading, not run on hardware
//
#include <hip/hip_runtime.h>

// ---------------------------------------------------------------------------
// Problem constants (from reference)
// ---------------------------------------------------------------------------
constexpr int BB    = 64;        // batch
constexpr int CIN   = 2048;
constexpr int COUT  = 32;
constexpr int HH    = 14;
constexpr int WW    = 14;
constexpr int HW    = 196;       // 14*14
constexpr int REP   = 1024;
constexpr int FEATN = CIN * COUT;   // 65536
constexpr int PSTR  = 200;          // padded HW rows in xt (rows 196.. are zeros)
constexpr int NSPLIT = 64;          // K-splits for the w6 GEMM
constexpr int KSP    = FEATN / NSPLIT; // 1024

// ---------------------------------------------------------------------------
// CDNA5 WMMA vector types / helpers (gfx1250, wave32)
// ---------------------------------------------------------------------------
typedef __attribute__((ext_vector_type(2)))  float  v2f;
typedef __attribute__((ext_vector_type(8)))  float  v8f;
typedef __attribute__((ext_vector_type(4)))  int    v4i;
typedef __attribute__((ext_vector_type(8)))  __bf16 v8bf;
typedef __attribute__((ext_vector_type(16))) __bf16 v16bf;

static __device__ __forceinline__ v8f wmma_f32x4(v2f a, v2f b, v8f c) {
  // V_WMMA_F32_16X16X4_F32 : exact f32 path for the bandwidth-bound GEMMs
  return __builtin_amdgcn_wmma_f32_16x16x4_f32(false, a, false, b, (short)0, c,
                                               false, false);
}
static __device__ __forceinline__ v8f wmma_bf16x32(v16bf a, v16bf b, v8f c) {
  // V_WMMA_F32_16X16X32_BF16 : high-throughput path for the compute-heavy conv
  return __builtin_amdgcn_wmma_f32_16x16x32_bf16(false, a, false, b, (short)0, c,
                                                 false, false);
}
static __device__ __forceinline__ v16bf cat16(v8bf lo, v8bf hi) {
  return __builtin_shufflevector(lo, hi, 0, 1, 2, 3, 4, 5, 6, 7,
                                         8, 9, 10, 11, 12, 13, 14, 15);
}

// ---------------------------------------------------------------------------
// CDNA5 async Global->LDS path (ASYNCcnt), feature-guarded so a toolchain
// without the builtin falls back to direct VMEM loads.
// Probe result (round 2 stderr): param 0 is `v4i addrspace(1)*` (prints as
// `int __vector(4) __device__ *`), so the builtin takes typed int4 pointers.
// ---------------------------------------------------------------------------
#if __has_builtin(__builtin_amdgcn_global_load_async_to_lds_b128)
#define USE_ASYNC_LDS 1
#else
#define USE_ASYNC_LDS 0
#endif

#if USE_ASYNC_LDS
static __device__ __forceinline__ void async_copy16(const float* gp, float* lp) {
  __builtin_amdgcn_global_load_async_to_lds_b128(
      (__attribute__((address_space(1))) v4i*)(void*)gp,
      (__attribute__((address_space(3))) v4i*)(void*)lp, 0, 0);
}
template <int N>
static __device__ __forceinline__ void wait_asynccnt() {
#if __has_builtin(__builtin_amdgcn_s_wait_asynccnt)
  __builtin_amdgcn_s_wait_asynccnt((short)N);
  asm volatile("" ::: "memory");
#else
  asm volatile("s_wait_asynccnt %0" ::"n"(N) : "memory");
#endif
}
#endif

// ---------------------------------------------------------------------------
// 1) conv_w [COUT][CIN][3][3] f32 -> wbf [tap][COUT][CIN] bf16 (tap-major)
//    Makes the WMMA A-fragment two aligned 16B loads per lane.
// ---------------------------------------------------------------------------
__global__ __launch_bounds__(256) void pack_w_kernel(const float* __restrict__ cw,
                                                     __bf16* __restrict__ wbf) {
  int t = blockIdx.x * blockDim.x + threadIdx.x;
  if (t >= 9 * COUT * CIN) return;
  int tap = t / (COUT * CIN);
  int rem = t - tap * (COUT * CIN);
  int o = rem >> 11;          // /CIN
  int i = rem & (CIN - 1);
  wbf[t] = (__bf16)cw[((size_t)o * CIN + i) * 9 + tap];
}

// ---------------------------------------------------------------------------
// 2) x [B][CIN][14][14] f32 -> xt [B][PSTR][CIN] bf16, rows 196..199 zeroed.
//    LDS 32x33 tile transpose: coalesced reads along hw, coalesced writes
//    along channel.
// ---------------------------------------------------------------------------
__global__ __launch_bounds__(256) void transpose_x_kernel(const float* __restrict__ x,
                                                          __bf16* __restrict__ xt) {
  __shared__ float tile[32][33];
  const int t  = threadIdx.x;
  const int p0 = blockIdx.x * 32;   // hw tile (7 tiles cover PSTR=200)
  const int i0 = blockIdx.y * 32;   // channel tile
  const int b  = blockIdx.z;
#pragma unroll
  for (int r = 0; r < 4; ++r) {
    int idx = t + 256 * r;          // 0..1023
    int ip = idx & 31, ii = idx >> 5;
    int p = p0 + ip;
    tile[ii][ip] = (p < HW) ? x[((size_t)b * CIN + (i0 + ii)) * HW + p] : 0.0f;
  }
  __syncthreads();
#pragma unroll
  for (int r = 0; r < 4; ++r) {
    int idx = t + 256 * r;
    int ii = idx & 31, ip = idx >> 5;
    int p = p0 + ip;
    if (p < PSTR)
      xt[((size_t)b * PSTR + p) * CIN + (i0 + ii)] = (__bf16)tile[ii][ip];
  }
}

// ---------------------------------------------------------------------------
// 3) Conv3x3 as 9 accumulated GEMMs on bf16 WMMA.
//    Per wave: one 16(M=cout) x 16(N=hw) tile, K = 9*2048 in 16x16x32 steps.
//    Out-of-image taps read the zeroed padding row 196 of xt (no divergence,
//    EXEC stays all-ones for WMMA).  grid(13, 64) block 64 (2 waves).
// ---------------------------------------------------------------------------
__global__ __launch_bounds__(64) void conv_wmma_kernel(const float* __restrict__ convb,
                                                       const __bf16* __restrict__ wbf,
                                                       const __bf16* __restrict__ xt,
                                                       float* __restrict__ am) {
  const int lane = threadIdx.x & 31;
  const int mt   = threadIdx.x >> 5;     // 0..1 -> cout tile
  const int nt   = blockIdx.x;           // 0..12 -> hw tile
  const int b    = blockIdx.y;
  const int ln   = lane & 15;
  const int hi   = lane >> 4;            // lane half selects K sub-block
  const int o    = mt * 16 + ln;         // A row (cout)
  const int p    = nt * 16 + ln;         // B col (hw position)
  const bool pv  = (p < HW);
  const int ph = p / WW, pw = p % WW;

  v8f acc0 = {}, acc1 = {};
  for (int tap = 0; tap < 9; ++tap) {
    const int dy = tap / 3 - 1, dx = tap % 3 - 1;
    const int y = ph + dy, xx = pw + dx;
    const bool bv = pv && ((unsigned)y < (unsigned)HH) && ((unsigned)xx < (unsigned)WW);
    const int q = bv ? (y * WW + xx) : HW;   // HW=196 -> zero-filled pad row
    const __bf16* aw = wbf + ((size_t)(tap * COUT + o)) * CIN + hi * 8;
    const __bf16* bx = xt + ((size_t)b * PSTR + q) * CIN + hi * 8;
#pragma unroll 2
    for (int i0 = 0; i0 < CIN; i0 += 64) {
      // dual accumulators break the WMMA->WMMA RAW chain
      v16bf A0 = cat16(*(const v8bf*)(aw + i0),      *(const v8bf*)(aw + i0 + 16));
      v16bf B0 = cat16(*(const v8bf*)(bx + i0),      *(const v8bf*)(bx + i0 + 16));
      acc0 = wmma_bf16x32(A0, B0, acc0);
      v16bf A1 = cat16(*(const v8bf*)(aw + i0 + 32), *(const v8bf*)(aw + i0 + 48));
      v16bf B1 = cat16(*(const v8bf*)(bx + i0 + 32), *(const v8bf*)(bx + i0 + 48));
      acc1 = wmma_bf16x32(A1, B1, acc1);
    }
  }
  v8f acc = acc0 + acc1;
  if (pv) {
#pragma unroll
    for (int j = 0; j < 8; ++j) {
      int oj = mt * 16 + j + 8 * hi;   // D row m = j + 8*(lane>=16)
      am[((size_t)b * COUT + oj) * HW + p] = acc[j] + convb[oj];
    }
  }
}

// ---------------------------------------------------------------------------
// 4) pooled/feat: per batch GEMM  C[i][o] = (1/196) * sum_hw x[b,i,hw]*am[b,o,hw]
//    M=2048 (i), N=32 (o), K=196. Exact f32 WMMA (16x16x4).
//    grid(128, 64) block 64 (2 waves = 2 o-tiles).
// ---------------------------------------------------------------------------
__global__ __launch_bounds__(64) void einsum_wmma_kernel(const float* __restrict__ x,
                                                         const float* __restrict__ am,
                                                         float* __restrict__ feat) {
  const int lane = threadIdx.x & 31;
  const int ot   = threadIdx.x >> 5;   // 0..1
  const int mt   = blockIdx.x;         // 0..127  (i tile)
  const int b    = blockIdx.y;
  const int ln   = lane & 15;
  const int hi   = lane >> 4;
  const int irow = mt * 16 + ln;
  const int o    = ot * 16 + ln;

  const float* xrow = x  + ((size_t)b * CIN  + irow) * HW + 2 * hi;
  const float* arow = am + ((size_t)b * COUT + o)    * HW + 2 * hi;
  v8f acc = {};
#pragma unroll 7
  for (int k = 0; k < HW; k += 4) {    // 49 exact steps
    v2f af = *(const v2f*)(xrow + k);
    v2f bf = *(const v2f*)(arow + k);
    acc = wmma_f32x4(af, bf, acc);
  }
  constexpr float inv = 1.0f / (float)HW;
#pragma unroll
  for (int j = 0; j < 8; ++j) {
    int i = mt * 16 + j + 8 * hi;
    feat[((size_t)b * CIN + i) * COUT + ot * 16 + ln] = acc[j] * inv;
  }
}

// ---------------------------------------------------------------------------
// 5) h6 partials: feat[64][65536] @ w6[1024][65536]^T, K split 64 ways.
//    THE bandwidth-critical kernel (268 MB of w6 -> ~11.5 us at 23.3 TB/s).
//    w6 is staged through LDS with double-buffered GLOBAL_LOAD_ASYNC_TO_LDS_B128:
//    one async op copies a full 512B row chunk (32 lanes x 16B contiguous);
//    16 ops stage a 16-row x 128-float chunk. B-fragments are then ds_load_b64
//    reads from LDS (row stride 132 floats -> 32 distinct banks per step).
//    s_wait_asynccnt 16/0 sequences the double buffer (in-order completion).
//    feat A-fragments stay direct (tiny, reused 64x -> L2-hot).
//    One wave per (n-tile, K-split) computes all 4 batch M-tiles so each
//    staged w6 fragment is reused 4x from registers.
//    grid(64, 64) block 32. Deterministic partials -> ws.
// ---------------------------------------------------------------------------
__global__ __launch_bounds__(32) void gemm6_stage1_kernel(const float* __restrict__ feat,
                                                          const float* __restrict__ w6,
                                                          float* __restrict__ part) {
  const int lane  = threadIdx.x;
  const int ln    = lane & 15;
  const int hi    = lane >> 4;
  const int nt    = blockIdx.x;     // 0..63 -> REP tile
  const int split = blockIdx.y;     // 0..63 -> K chunk
  const int r     = nt * 16 + ln;

  const float* f0r = feat + (size_t)ln * FEATN + (size_t)split * KSP + 2 * hi;
  const float* f1r = f0r + (size_t)16 * FEATN;
  const float* f2r = f0r + (size_t)32 * FEATN;
  const float* f3r = f0r + (size_t)48 * FEATN;

  v8f a0 = {}, a1 = {}, a2 = {}, a3 = {};

#if USE_ASYNC_LDS
  // 16 rows x 128 floats per chunk, row padded to 132 floats for bank spread.
  __shared__ float sw[2][16][132];
  const float* gbase = w6 + (size_t)(nt * 16) * FEATN + (size_t)split * KSP;
  const int seg = lane * 4;          // 16B segment within a 512B row chunk

#define STAGE_CHUNK(bufi, k0s)                                              \
  do {                                                                      \
    _Pragma("unroll")                                                       \
    for (int i = 0; i < 16; ++i)                                            \
      async_copy16(gbase + (size_t)i * FEATN + (k0s) + seg,                 \
                   &sw[(bufi)][i][seg]);                                    \
  } while (0)

  STAGE_CHUNK(0, 0);
  for (int c = 0; c < 8; ++c) {
    if (c < 7) {
      STAGE_CHUNK((c + 1) & 1, (c + 1) * 128);
      wait_asynccnt<16>();           // chunk c complete (in-order), c+1 in flight
    } else {
      wait_asynccnt<0>();
    }
    const int buf = c & 1;
    const int k0  = c * 128;
#pragma unroll 8
    for (int kk = 0; kk < 128; kk += 4) {
      v2f bf = *(const v2f*)&sw[buf][ln][kk + 2 * hi];
      const int k = k0 + kk;
      a0 = wmma_f32x4(*(const v2f*)(f0r + k), bf, a0);
      a1 = wmma_f32x4(*(const v2f*)(f1r + k), bf, a1);
      a2 = wmma_f32x4(*(const v2f*)(f2r + k), bf, a2);
      a3 = wmma_f32x4(*(const v2f*)(f3r + k), bf, a3);
    }
  }
#undef STAGE_CHUNK
#else
  // Fallback: direct VMEM streaming of w6 (contiguous 16B/row/step) + prefetch.
  const float* wrow = w6 + (size_t)r * FEATN + (size_t)split * KSP + 2 * hi;
#pragma unroll 4
  for (int k = 0; k < KSP; k += 4) {
    if ((k & 127) == 0) __builtin_prefetch(wrow + k + 256, 0, 0);
    v2f bf = *(const v2f*)(wrow + k);
    a0 = wmma_f32x4(*(const v2f*)(f0r + k), bf, a0);
    a1 = wmma_f32x4(*(const v2f*)(f1r + k), bf, a1);
    a2 = wmma_f32x4(*(const v2f*)(f2r + k), bf, a2);
    a3 = wmma_f32x4(*(const v2f*)(f3r + k), bf, a3);
  }
#endif

  v8f* accs[4] = {&a0, &a1, &a2, &a3};
#pragma unroll
  for (int m = 0; m < 4; ++m) {
#pragma unroll
    for (int j = 0; j < 8; ++j) {
      int bb = m * 16 + j + 8 * hi;
      part[((size_t)split * BB + bb) * REP + r] = (*accs[m])[j];
    }
  }
}

// ---------------------------------------------------------------------------
// 6) reduce partials over splits, + bias, ReLU -> h6[64][1024]
// ---------------------------------------------------------------------------
__global__ __launch_bounds__(256) void reduce6_kernel(const float* __restrict__ part,
                                                      const float* __restrict__ b6,
                                                      float* __restrict__ h6) {
  int t = blockIdx.x * blockDim.x + threadIdx.x;   // 65536 threads
  int b = t >> 10, r = t & (REP - 1);
  float s = 0.0f;
#pragma unroll 8
  for (int sp = 0; sp < NSPLIT; ++sp)
    s += part[((size_t)sp * BB + b) * REP + r];
  h6[t] = fmaxf(s + b6[r], 0.0f);
}

// ---------------------------------------------------------------------------
// 7) out = relu(h6 @ w7^T + b7) : M=64, N=1024, K=1024. f32 WMMA.
//    grid 64, block 32; one wave per n-tile handles all 4 batch tiles.
// ---------------------------------------------------------------------------
__global__ __launch_bounds__(32) void gemm7_kernel(const float* __restrict__ h6,
                                                   const float* __restrict__ w7,
                                                   const float* __restrict__ b7,
                                                   float* __restrict__ out) {
  const int lane = threadIdx.x;
  const int ln   = lane & 15;
  const int hi   = lane >> 4;
  const int nt   = blockIdx.x;
  const int r    = nt * 16 + ln;

  const float* wrow = w7 + (size_t)r  * REP + 2 * hi;
  const float* h0r  = h6 + (size_t)ln * REP + 2 * hi;
  const float* h1r  = h0r + (size_t)16 * REP;
  const float* h2r  = h0r + (size_t)32 * REP;
  const float* h3r  = h0r + (size_t)48 * REP;

  v8f a0 = {}, a1 = {}, a2 = {}, a3 = {};
#pragma unroll 4
  for (int k = 0; k < REP; k += 4) {
    v2f bf = *(const v2f*)(wrow + k);
    a0 = wmma_f32x4(*(const v2f*)(h0r + k), bf, a0);
    a1 = wmma_f32x4(*(const v2f*)(h1r + k), bf, a1);
    a2 = wmma_f32x4(*(const v2f*)(h2r + k), bf, a2);
    a3 = wmma_f32x4(*(const v2f*)(h3r + k), bf, a3);
  }
  const float bias = b7[r];
  v8f* accs[4] = {&a0, &a1, &a2, &a3};
#pragma unroll
  for (int m = 0; m < 4; ++m) {
#pragma unroll
    for (int j = 0; j < 8; ++j) {
      int bb = m * 16 + j + 8 * hi;
      out[(size_t)bb * REP + r] = fmaxf((*accs[m])[j] + bias, 0.0f);
    }
  }
}

// ---------------------------------------------------------------------------
// Workspace layout (bytes; every region fully written each call -> no init):
//   wbf  : 9*32*2048       bf16 =  1,179,648
//   xt   : 64*200*2048     bf16 = 52,428,800
//   am   : 64*32*196       f32  =  1,605,632
//   feat : 64*65536        f32  = 16,777,216
//   part : 64*64*1024      f32  = 16,777,216
//   h6   : 64*1024         f32  =     262,144
//   total ~ 85 MB
// ---------------------------------------------------------------------------
constexpr size_t WBF_BYTES  = (size_t)9 * COUT * CIN * 2;
constexpr size_t XT_BYTES   = (size_t)BB * PSTR * CIN * 2;
constexpr size_t AM_BYTES   = (size_t)BB * COUT * HW * 4;
constexpr size_t FEAT_BYTES = (size_t)BB * FEATN * 4;
constexpr size_t PART_BYTES = (size_t)NSPLIT * BB * REP * 4;

extern "C" void kernel_launch(void* const* d_in, const int* in_sizes, int n_in,
                              void* d_out, int out_size, void* d_ws, size_t ws_size,
                              hipStream_t stream) {
  const float* x      = (const float*)d_in[0];
  const float* conv_w = (const float*)d_in[1];
  const float* conv_b = (const float*)d_in[2];
  const float* w6     = (const float*)d_in[3];
  const float* b6     = (const float*)d_in[4];
  const float* w7     = (const float*)d_in[5];
  const float* b7     = (const float*)d_in[6];
  float* out = (float*)d_out;

  char* ws = (char*)d_ws;
  __bf16* wbf  = (__bf16*)(ws);
  __bf16* xt   = (__bf16*)(ws + WBF_BYTES);
  float*  am   = (float*) (ws + WBF_BYTES + XT_BYTES);
  float*  feat = (float*) (ws + WBF_BYTES + XT_BYTES + AM_BYTES);
  float*  part = (float*) (ws + WBF_BYTES + XT_BYTES + AM_BYTES + FEAT_BYTES);
  float*  h6   = (float*) (ws + WBF_BYTES + XT_BYTES + AM_BYTES + FEAT_BYTES + PART_BYTES);

  pack_w_kernel<<<(9 * COUT * CIN + 255) / 256, 256, 0, stream>>>(conv_w, wbf);
  transpose_x_kernel<<<dim3(7, CIN / 32, BB), 256, 0, stream>>>(x, xt);
  conv_wmma_kernel<<<dim3(13, BB), 64, 0, stream>>>(conv_b, wbf, xt, am);
  einsum_wmma_kernel<<<dim3(CIN / 16, BB), 64, 0, stream>>>(x, am, feat);
  gemm6_stage1_kernel<<<dim3(REP / 16, NSPLIT), 32, 0, stream>>>(feat, w6, part);
  reduce6_kernel<<<(BB * REP) / 256, 256, 0, stream>>>(part, b6, h6);
  gemm7_kernel<<<REP / 16, 32, 0, stream>>>(h6, w7, b7, out);
}
